// ContrastiveSWM_1623497638135
// MI455X (gfx1250) — matrix operand order repaired and checked by reading, not compiled, not run on hardware
//
#include <hip/hip_runtime.h>

typedef __bf16 bf16;
typedef __attribute__((ext_vector_type(16))) __bf16 v16bf;
typedef __attribute__((ext_vector_type(8)))  __bf16 v8bf;
typedef __attribute__((ext_vector_type(8)))  float  v8f;

#define FLAG_LN    1
#define FLAG_RELU  2
#define FLAG_OUTB  4
#define FLAG_OUTF  8
#define FLAG_RESID 16

// ---------------------------------------------------------------------------
// Fused GEMM: out = act(LN(A @ W + bias)). A (M x K) bf16 row-major, WT (N x K)
// bf16 pre-transposed. Workgroup = 256 threads = 8 wave32 waves covering
// 64 rows x N (full width, so LayerNorm reduces inside the WG). N = NTILES*128
// is compile-time; FLAGS is compile-time -> branch-free double-buffered K loop.
// ---------------------------------------------------------------------------
template<int NTILES, int FLAGS>
__global__ __launch_bounds__(256) void swm_gemm(
    const bf16* __restrict__ A, const bf16* __restrict__ WT,
    const float* __restrict__ bias, const float* __restrict__ gamma,
    const float* __restrict__ beta,
    bf16* __restrict__ outb, float* __restrict__ outf,
    const float* __restrict__ resid, int K)
{
    constexpr int N = NTILES * 128;
    __shared__ float s_sum[64];
    __shared__ float s_ssq[64];

    const int tid  = threadIdx.x;
    const int wave = tid >> 5;
    const int lane = tid & 31;
    const int h    = lane >> 4;      // half of wave (0: lanes 0-15, 1: 16-31)
    const int lr   = lane & 15;
    const int colbase = wave * (N >> 3);
    const long rowbase = (long)blockIdx.x * 64;

    if constexpr (FLAGS & FLAG_LN) {
        if (tid < 64) { s_sum[tid] = 0.f; s_ssq[tid] = 0.f; }
        __syncthreads();
    }

    // Per-fragment base pointers (ISA 16-bit A / B VGPR layouts).
    const bf16* arow[4];
    #pragma unroll
    for (int m = 0; m < 4; ++m)
        arow[m] = A + (size_t)(rowbase + m * 16 + lr) * K + h * 8;
    const bf16* brow[NTILES];
    #pragma unroll
    for (int t = 0; t < NTILES; ++t)
        brow[t] = WT + (size_t)(colbase + t * 16 + lr) * K + h * 8;

    v8f acc[4][NTILES];
    #pragma unroll
    for (int m = 0; m < 4; ++m)
        #pragma unroll
        for (int t = 0; t < NTILES; ++t)
            #pragma unroll
            for (int j = 0; j < 8; ++j)
                acc[m][t][j] = 0.f;

    auto loadstep = [&](v16bf (&af)[4], v16bf (&bw)[NTILES], int k0) {
        #pragma unroll
        for (int m = 0; m < 4; ++m) {
            v8bf alo = *(const v8bf*)(arow[m] + k0);
            v8bf ahi = *(const v8bf*)(arow[m] + k0 + 16);
            af[m] = __builtin_shufflevector(alo, ahi,
                0,1,2,3,4,5,6,7,8,9,10,11,12,13,14,15);
            __builtin_prefetch(arow[m] + k0 + 64, 0, 1);  // stream A ahead
        }
        #pragma unroll
        for (int t = 0; t < NTILES; ++t)
            bw[t] = __builtin_shufflevector(*(const v8bf*)(brow[t] + k0),
                *(const v8bf*)(brow[t] + k0 + 16),
                0,1,2,3,4,5,6,7,8,9,10,11,12,13,14,15);
    };
    auto mmastep = [&](const v16bf (&af)[4], const v16bf (&bw)[NTILES]) {
        #pragma unroll
        for (int t = 0; t < NTILES; ++t)
            #pragma unroll
            for (int m = 0; m < 4; ++m)
                acc[m][t] = __builtin_amdgcn_wmma_f32_16x16x32_bf16(
                    false, af[m], false, bw[t], (short)0, acc[m][t],
                    false, false);
    };

    // Double-buffered K loop: fetch step s+1 while the 16 WMMAs of step s run.
    const int steps = K >> 5;
    v16bf aA[4], bA[NTILES], aB[4], bB[NTILES];
    loadstep(aA, bA, 0);
    int s = 0;
    for (; s + 2 <= steps; s += 2) {
        loadstep(aB, bB, (s + 1) << 5);
        mmastep(aA, bA);
        if (s + 2 < steps) loadstep(aA, bA, (s + 2) << 5);
        mmastep(aB, bB);
    }
    if (s < steps) mmastep(aA, bA);   // odd tail (e.g. K=672 -> 21 steps)

    // Bias + per-column LN params (hoisted).
    float bc[NTILES], gc[NTILES], btc[NTILES];
    #pragma unroll
    for (int t = 0; t < NTILES; ++t) {
        const int col = colbase + t * 16 + lr;
        bc[t] = bias[col];
        if constexpr (FLAGS & FLAG_LN) { gc[t] = gamma[col]; btc[t] = beta[col]; }
    }
    #pragma unroll
    for (int m = 0; m < 4; ++m)
        #pragma unroll
        for (int t = 0; t < NTILES; ++t)
            #pragma unroll
            for (int j = 0; j < 8; ++j)
                acc[m][t][j] += bc[t];

    // Fused LayerNorm statistics: per row (m*16 + j + 8*h) reduce the wave's
    // N/8 columns with shuffles, then cross-wave via LDS float atomics.
    if constexpr (FLAGS & FLAG_LN) {
        #pragma unroll
        for (int m = 0; m < 4; ++m) {
            #pragma unroll
            for (int j = 0; j < 8; ++j) {
                float s1 = 0.f, s2 = 0.f;
                #pragma unroll
                for (int t = 0; t < NTILES; ++t) {
                    const float x = acc[m][t][j];
                    s1 += x; s2 += x * x;
                }
                #pragma unroll
                for (int off = 1; off < 16; off <<= 1) {
                    s1 += __shfl_xor(s1, off, 32);
                    s2 += __shfl_xor(s2, off, 32);
                }
                if (lr == 0) {
                    const int r = m * 16 + j + 8 * h;
                    atomicAdd(&s_sum[r], s1);
                    atomicAdd(&s_ssq[r], s2);
                }
            }
        }
        __syncthreads();
    }

    // Epilogue: LN -> ReLU -> store (bf16 activation and/or f32 + residual).
    constexpr float invN = 1.0f / (float)N;
    #pragma unroll
    for (int m = 0; m < 4; ++m) {
        #pragma unroll
        for (int j = 0; j < 8; ++j) {
            const int  r    = m * 16 + j + 8 * h;
            const long grow = rowbase + r;
            float mu = 0.f, rs = 0.f;
            if constexpr (FLAGS & FLAG_LN) {
                mu = s_sum[r] * invN;
                const float var = s_ssq[r] * invN - mu * mu;
                rs = rsqrtf(var + 1e-5f);
            }
            #pragma unroll
            for (int t = 0; t < NTILES; ++t) {
                const int col = colbase + t * 16 + lr;
                float x = acc[m][t][j];
                if constexpr (FLAGS & FLAG_LN)   x = (x - mu) * rs * gc[t] + btc[t];
                if constexpr (FLAGS & FLAG_RELU) x = fmaxf(x, 0.f);
                const size_t idx = (size_t)grow * N + col;
                if constexpr (FLAGS & FLAG_OUTB) outb[idx] = (bf16)x;
                if constexpr (FLAGS & FLAG_OUTF) {
                    if constexpr (FLAGS & FLAG_RESID) outf[idx] = x + resid[idx];
                    else                              outf[idx] = x;
                }
            }
        }
    }
}

// --- weight transpose+pad: src f32 (K x N) -> dst bf16 (N x Kpad), zeros pad
__global__ void swm_wtrans(const float* __restrict__ src, bf16* __restrict__ dst,
                           int K, int N, int Kpad, int total)
{
    const int i = blockIdx.x * 256 + threadIdx.x;
    if (i >= total) return;
    const int n = i / Kpad, k = i % Kpad;
    dst[i] = (k < K) ? (bf16)src[(size_t)k * N + n] : (bf16)0.f;
}

// --- obs (B x 10 f32) -> A0 (B x 32 bf16, zero-padded)
__global__ void swm_obspad(const float* __restrict__ obs, bf16* __restrict__ dst,
                           int total)
{
    const int i = blockIdx.x * 256 + threadIdx.x;
    if (i >= total) return;
    const int b = i >> 5, j = i & 31;
    dst[i] = (j < 10) ? (bf16)obs[b * 10 + j] : (bf16)0.f;
}

// --- edge input gather: [state[b,ROWS[e]], state[b,COLS[e]]] -> (B*6) x 256
__global__ void swm_edgein(const bf16* __restrict__ stateb, bf16* __restrict__ ein,
                           int total)
{
    const int i = blockIdx.x * 256 + threadIdx.x;
    if (i >= total) return;
    const int d = i & 255;
    const int r = i >> 8;             // r = b*6 + e
    const int e = r % 6;
    const int b = r / 6;
    const int src  = e >> 1;          // ROWS = [0,0,1,1,2,2]
    const int j0   = (src == 0) ? 1 : 0;
    const int j1   = (src == 2) ? 1 : 2;
    const int tgt  = (e & 1) ? j1 : j0;   // COLS = [1,2,0,2,0,1]
    const int node = (d < 128) ? src : tgt;
    ein[i] = stateb[((size_t)b * 3 + node) * 128 + (d & 127)];
}

// --- aggregation: agg[b,i] = edge_out[b,2i] + edge_out[b,2i+1]
__global__ void swm_agg(const bf16* __restrict__ eo, bf16* __restrict__ agg,
                        int total)
{
    const int i = blockIdx.x * 256 + threadIdx.x;
    if (i >= total) return;
    const int d = i & 511;
    const int r = i >> 9;             // r = b*3 + node
    const int node = r % 3, b = r / 3;
    const size_t e0 = ((size_t)b * 6 + node * 2) * 512 + d;
    agg[i] = (bf16)((float)eo[e0] + (float)eo[e0 + 512]);
}

// --- node input concat: [state(128) | one_hot(action,5) | agg(512) | pad(27)]
__global__ void swm_nodein(const bf16* __restrict__ stateb,
                           const int* __restrict__ action,
                           const bf16* __restrict__ agg,
                           bf16* __restrict__ nin, int total)
{
    const int i = blockIdx.x * 256 + threadIdx.x;
    if (i >= total) return;
    const int k = i % 672;
    const int r = i / 672;            // r = b*3 + node
    bf16 v = (bf16)0.f;
    if (k < 128)      v = stateb[(size_t)r * 128 + k];
    else if (k < 133) v = (action[r] == (k - 128)) ? (bf16)1.f : (bf16)0.f;
    else if (k < 645) v = agg[(size_t)r * 512 + (k - 133)];
    nin[i] = v;
}

extern "C" void kernel_launch(void* const* d_in, const int* in_sizes, int n_in,
                              void* d_out, int out_size, void* d_ws, size_t ws_size,
                              hipStream_t stream)
{
    (void)n_in; (void)out_size; (void)ws_size;
    const int B = in_sizes[0] / 10;           // 16384
    const long M1 = B, M6 = 6L * B, M3 = 3L * B;

    const float* obs    = (const float*)d_in[0];
    const int*   action = (const int*)d_in[1];
    auto F = [&](int i) { return (const float*)d_in[i]; };
    // enc: w1 b1 g1 bb1 w2 b2 g2 bb2 w3 b3 -> d_in[2..11]
    // edge: w0 b0 (w b g beta)x2 wf bf     -> d_in[12..23]
    // node: w0 b0 (w b g beta)x2 wf bf     -> d_in[24..35]

    char* ws = (char*)d_ws;
    size_t off = 0;
    auto take = [&](size_t elems, size_t esz) -> char* {
        char* p = ws + off;
        off += (elems * esz + 255) & ~(size_t)255;
        return p;
    };
    // transposed bf16 weights
    bf16* w1T  = (bf16*)take(512 * 32, 2);
    bf16* w2T  = (bf16*)take(512 * 512, 2);
    bf16* w3T  = (bf16*)take(384 * 512, 2);
    bf16* ew0T = (bf16*)take(512 * 256, 2);
    bf16* eh0T = (bf16*)take(512 * 512, 2);
    bf16* eh1T = (bf16*)take(512 * 512, 2);
    bf16* ewfT = (bf16*)take(512 * 512, 2);
    bf16* nw0T = (bf16*)take(512 * 672, 2);
    bf16* nh0T = (bf16*)take(512 * 512, 2);
    bf16* nh1T = (bf16*)take(512 * 512, 2);
    bf16* nwfT = (bf16*)take(128 * 512, 2);
    // activations
    bf16*  A0   = (bf16*)take((size_t)M1 * 32, 2);
    bf16*  H1   = (bf16*)take((size_t)M1 * 512, 2);
    bf16*  H2   = (bf16*)take((size_t)M1 * 512, 2);
    bf16*  SBF  = (bf16*)take((size_t)M1 * 384, 2);
    float* SF32 = (float*)take((size_t)M1 * 384, 4);
    bf16*  AGG  = (bf16*)take((size_t)M3 * 512, 2);
    bf16*  EIN  = (bf16*)take((size_t)M6 * 256, 2);
    bf16*  EB1  = (bf16*)take((size_t)M6 * 512, 2);   // also node ping
    bf16*  EB2  = (bf16*)take((size_t)M6 * 512, 2);   // also node-input / pong

    auto wt = [&](int idx, bf16* dst, int K, int N, int Kpad) {
        const int total = N * Kpad;
        swm_wtrans<<<dim3((total + 255) / 256), dim3(256), 0, stream>>>(
            F(idx), dst, K, N, Kpad, total);
    };
    wt(2,  w1T,  10,  512, 32);
    wt(6,  w2T,  512, 512, 512);
    wt(10, w3T,  512, 384, 512);
    wt(12, ew0T, 256, 512, 256);
    wt(14, eh0T, 512, 512, 512);
    wt(18, eh1T, 512, 512, 512);
    wt(22, ewfT, 512, 512, 512);
    wt(24, nw0T, 645, 512, 672);
    wt(26, nh0T, 512, 512, 512);
    wt(30, nh1T, 512, 512, 512);
    wt(34, nwfT, 512, 128, 512);

    // encoder
    {
        const int total = B * 32;
        swm_obspad<<<dim3((total + 255) / 256), dim3(256), 0, stream>>>(obs, A0, total);
    }
    swm_gemm<4, FLAG_LN | FLAG_RELU | FLAG_OUTB>
        <<<dim3((unsigned)(M1 / 64)), dim3(256), 0, stream>>>(
            A0, w1T, F(3), F(4), F(5), H1, nullptr, nullptr, 32);
    swm_gemm<4, FLAG_LN | FLAG_RELU | FLAG_OUTB>
        <<<dim3((unsigned)(M1 / 64)), dim3(256), 0, stream>>>(
            H1, w2T, F(7), F(8), F(9), H2, nullptr, nullptr, 512);
    swm_gemm<3, FLAG_OUTB | FLAG_OUTF>
        <<<dim3((unsigned)(M1 / 64)), dim3(256), 0, stream>>>(
            H2, w3T, F(11), nullptr, nullptr, SBF, SF32, nullptr, 512);

    // edge MLP
    {
        const long total = M6 * 256;
        swm_edgein<<<dim3((unsigned)((total + 255) / 256)), dim3(256), 0, stream>>>(
            SBF, EIN, (int)total);
    }
    swm_gemm<4, FLAG_RELU | FLAG_OUTB>
        <<<dim3((unsigned)(M6 / 64)), dim3(256), 0, stream>>>(
            EIN, ew0T, F(13), nullptr, nullptr, EB1, nullptr, nullptr, 256);
    swm_gemm<4, FLAG_LN | FLAG_RELU | FLAG_OUTB>
        <<<dim3((unsigned)(M6 / 64)), dim3(256), 0, stream>>>(
            EB1, eh0T, F(15), F(16), F(17), EB2, nullptr, nullptr, 512);
    swm_gemm<4, FLAG_LN | FLAG_RELU | FLAG_OUTB>
        <<<dim3((unsigned)(M6 / 64)), dim3(256), 0, stream>>>(
            EB2, eh1T, F(19), F(20), F(21), EB1, nullptr, nullptr, 512);
    swm_gemm<4, FLAG_OUTB>
        <<<dim3((unsigned)(M6 / 64)), dim3(256), 0, stream>>>(
            EB1, ewfT, F(23), nullptr, nullptr, EB2, nullptr, nullptr, 512);

    // aggregation + node input
    {
        const long total = M3 * 512;
        swm_agg<<<dim3((unsigned)((total + 255) / 256)), dim3(256), 0, stream>>>(
            EB2, AGG, (int)total);
    }
    bf16* NIN = EB2;   // reuse (edge_out already consumed by swm_agg)
    {
        const long total = M3 * 672;
        swm_nodein<<<dim3((unsigned)((total + 255) / 256)), dim3(256), 0, stream>>>(
            SBF, action, AGG, NIN, (int)total);
    }

    // node MLP (+ residual into d_out)
    swm_gemm<4, FLAG_RELU | FLAG_OUTB>
        <<<dim3((unsigned)(M3 / 64)), dim3(256), 0, stream>>>(
            NIN, nw0T, F(25), nullptr, nullptr, EB1, nullptr, nullptr, 672);
    swm_gemm<4, FLAG_LN | FLAG_RELU | FLAG_OUTB>
        <<<dim3((unsigned)(M3 / 64)), dim3(256), 0, stream>>>(
            EB1, nh0T, F(27), F(28), F(29), EB2, nullptr, nullptr, 512);
    swm_gemm<4, FLAG_LN | FLAG_RELU | FLAG_OUTB>
        <<<dim3((unsigned)(M3 / 64)), dim3(256), 0, stream>>>(
            EB2, nh1T, F(31), F(32), F(33), EB1, nullptr, nullptr, 512);
    swm_gemm<1, FLAG_OUTF | FLAG_RESID>
        <<<dim3((unsigned)(M3 / 64)), dim3(256), 0, stream>>>(
            EB1, nwfT, F(35), nullptr, nullptr, nullptr, (float*)d_out, SF32, 512);
}
